// Sc_encoder_41437844471882
// MI455X (gfx1250) — compile-verified, matrix operand order checked by reading, not run on hardware
//
#include <hip/hip_runtime.h>

#define N_NODES 8192
#define DIM 64
#define S_NEI 50
#define INV_TAU 2.0f

typedef __attribute__((ext_vector_type(16))) _Float16 v16h;
typedef __attribute__((ext_vector_type(8)))  float    v8f;
typedef __attribute__((ext_vector_type(4)))  float    vf4;
typedef __attribute__((ext_vector_type(2)))  float    vf2;

union Frag { v16h h; vf4 f[2]; };

__device__ __forceinline__ float wred32(float v) {
  v += __shfl_xor(v, 16, 32);
  v += __shfl_xor(v, 8, 32);
  v += __shfl_xor(v, 4, 32);
  v += __shfl_xor(v, 2, 32);
  v += __shfl_xor(v, 1, 32);
  return v;
}

// A/B fragment load for v_wmma_f32_16x16x32_f16.
// Lane (g = lane>>4, m = lane&15) holds row m of the 16x32 tile:
// halves[0..7]  = row[kblk*32 + 8g .. +7]
// halves[8..15] = row[kblk*32 + 16 + 8g .. +7]
__device__ __forceinline__ v16h load_frag(const _Float16* base, int row, int g, int kblk) {
  const _Float16* p = base + row * DIM + kblk * 32;
  Frag fr;
  fr.f[0] = *(const vf4*)(p + 8 * g);
  fr.f[1] = *(const vf4*)(p + 16 + 8 * g);
  return fr.h;
}

// ---------------- K1: intra-type GAT attention (one wave per node) ----------
__global__ __launch_bounds__(256) void k1_intra(
    const float* __restrict__ h0, const float* __restrict__ h1,
    const float* __restrict__ h2, const int* __restrict__ nei0,
    const int* __restrict__ nei1, const float* __restrict__ att0,
    const float* __restrict__ att1, float* __restrict__ e0,
    float* __restrict__ e1) {
  const int t = blockIdx.y;
  const float* h   = t ? h2   : h1;
  const int*   nei = t ? nei1 : nei0;
  const float* att = t ? att1 : att0;
  float*       out = t ? e1   : e0;

  const int wave = threadIdx.x >> 5, lane = threadIdx.x & 31;
  const int n = blockIdx.x * 8 + wave;

  vf2 hr = *(const vf2*)(h0 + n * DIM + 2 * lane);
  vf2 ar = *(const vf2*)(att + 2 * lane);
  vf2 an = *(const vf2*)(att + DIM + 2 * lane);
  float base = wred32(hr.x * ar.x + hr.y * ar.y);

  float mx = -1e30f, ssum = 0.f;
  float ax = 0.f, ay = 0.f;
  const int* np = nei + n * S_NEI;
  for (int s = 0; s < S_NEI; ++s) {
    int idx = np[s];
    vf2 v = *(const vf2*)(h + idx * DIM + 2 * lane);
    float logit = base + wred32(v.x * an.x + v.y * an.y);
    logit = logit > 0.f ? logit : 0.01f * logit;       // leaky_relu(0.01)
    float nm = fmaxf(mx, logit);
    float sc = __expf(mx - nm);
    float w  = __expf(logit - nm);
    ssum = ssum * sc + w;
    ax = ax * sc + w * v.x;
    ay = ay * sc + w * v.y;
    mx = nm;
  }
  float inv = 1.f / ssum;
  float o0 = ax * inv, o1 = ay * inv;
  o0 = o0 > 0.f ? o0 : expm1f(o0);                     // elu
  o1 = o1 > 0.f ? o1 : expm1f(o1);
  vf2 o; o.x = o0; o.y = o1;
  *(vf2*)(out + n * DIM + 2 * lane) = o;
}

// ---------------- K2: row-normalize, emit f16 + inverse norms ---------------
__global__ __launch_bounds__(256) void k2_norm(
    const float* __restrict__ h0, const float* __restrict__ e0,
    const float* __restrict__ e1, _Float16* __restrict__ a16,
    _Float16* __restrict__ b16, _Float16* __restrict__ c16,
    float* __restrict__ invn) {
  const int which = blockIdx.y;
  const float* src = which == 0 ? h0 : (which == 1 ? e0 : e1);
  _Float16*    dst = which == 0 ? a16 : (which == 1 ? b16 : c16);
  const int wave = threadIdx.x >> 5, lane = threadIdx.x & 31;
  const int n = blockIdx.x * 8 + wave;
  vf2 v = *(const vf2*)(src + n * DIM + 2 * lane);
  float inv = rsqrtf(wred32(v.x * v.x + v.y * v.y));
  if (lane == 0) invn[which * N_NODES + n] = inv;
  dst[n * DIM + 2 * lane]     = (_Float16)(v.x * inv);
  dst[n * DIM + 2 * lane + 1] = (_Float16)(v.y * inv);
}

// ---------------- K2b: exact fp32 diagonal similarity sums ------------------
__global__ __launch_bounds__(256) void k2b_diag(
    const float* __restrict__ h0, const float* __restrict__ e0,
    const float* __restrict__ e1, const float* __restrict__ invn,
    float* __restrict__ acc) {
  __shared__ float bsum;
  const int p = blockIdx.y;                  // 0:(h0,e0) 1:(h0,e1) 2:(e0,e1)
  const float* X = (p == 2) ? e0 : h0;
  const float* Y = (p == 0) ? e0 : e1;
  const int xi = (p == 2) ? 1 : 0;
  const int yi = (p == 0) ? 1 : 2;
  if (threadIdx.x == 0) bsum = 0.f;
  __syncthreads();
  const int wave = threadIdx.x >> 5, lane = threadIdx.x & 31;
  const int n = blockIdx.x * 8 + wave;
  vf2 x = *(const vf2*)(X + n * DIM + 2 * lane);
  vf2 y = *(const vf2*)(Y + n * DIM + 2 * lane);
  float d = wred32(x.x * y.x + x.y * y.y);
  if (lane == 0) {
    d *= invn[xi * N_NODES + n] * invn[yi * N_NODES + n] * INV_TAU;
    atomicAdd(&bsum, d);
  }
  __syncthreads();
  if (threadIdx.x == 0) atomicAdd(&acc[p], bsum);
}

// ---------------- K3: WMMA sim tiles + exp, row/col sum accumulation --------
__global__ __launch_bounds__(256) void k3_sim(
    const _Float16* __restrict__ a16, const _Float16* __restrict__ b16,
    const _Float16* __restrict__ c16, float* __restrict__ rc) {
  const int p = blockIdx.z;
  const _Float16* X = (p == 2) ? b16 : a16;
  const _Float16* Y = (p == 0) ? b16 : c16;
  float* rowsum = rc + p * N_NODES;
  float* colsum = rc + (3 + p) * N_NODES;

  const int wave = threadIdx.x >> 5, lane = threadIdx.x & 31;
  const int g = lane >> 4, m = lane & 15;
  const int i0 = (blockIdx.x * 8 + wave) * 16;     // 16-row strip of X
  const int jbase = blockIdx.y * 1024;             // 1024-column chunk of Y

  v16h A0 = load_frag(X, i0 + m, g, 0);
  v16h A1 = load_frag(X, i0 + m, g, 1);

  float rs[8];
#pragma unroll
  for (int r = 0; r < 8; ++r) rs[r] = 0.f;

  for (int jt = 0; jt < 64; ++jt) {
    const int col = jbase + jt * 16;
    v16h B0 = load_frag(Y, col + m, g, 0);
    v16h B1 = load_frag(Y, col + m, g, 1);
    v8f cacc = {};
    cacc = __builtin_amdgcn_wmma_f32_16x16x32_f16(false, A0, false, B0,
                                                  (short)0, cacc, false, false);
    cacc = __builtin_amdgcn_wmma_f32_16x16x32_f16(false, A1, false, B1,
                                                  (short)0, cacc, false, false);
    float cp = 0.f;
#pragma unroll
    for (int r = 0; r < 8; ++r) {
      float e = __expf(cacc[r] * INV_TAU);  // sim in [-2,2]: no max shift needed
      rs[r] += e;
      cp += e;
    }
    cp += __shfl_xor(cp, 16, 32);           // combine the two 16-row halves
    if (lane < 16) atomicAdd(&colsum[col + m], cp);
  }
#pragma unroll
  for (int r = 0; r < 8; ++r) {             // reduce across N within 16-lane group
    float v = rs[r];
    v += __shfl_xor(v, 8, 32);
    v += __shfl_xor(v, 4, 32);
    v += __shfl_xor(v, 2, 32);
    v += __shfl_xor(v, 1, 32);
    if (m == 0) atomicAdd(&rowsum[i0 + r + 8 * g], v);
  }
}

// ---------------- K4: sps[t] = mean_n tanh(e_t @ fc_w^T + fc_b) -------------
__global__ __launch_bounds__(256) void k4_fc(
    const float* __restrict__ e0, const float* __restrict__ e1,
    const float* __restrict__ fc_w, const float* __restrict__ fc_b,
    float* __restrict__ acc) {
  __shared__ float lw[DIM * DIM];
  const int t = blockIdx.y;
  const float* e = t ? e1 : e0;
  for (int i = threadIdx.x; i < DIM * DIM; i += 256) lw[i] = fc_w[i];
  __syncthreads();
  const int wave = threadIdx.x >> 5, lane = threadIdx.x & 31;
  const int d0 = 2 * lane, d1 = 2 * lane + 1;
  const float b0 = fc_b[d0], b1 = fc_b[d1];
  float s0 = 0.f, s1 = 0.f;
  const int r0 = (blockIdx.x * 8 + wave) * 16;
  for (int i = 0; i < 16; ++i) {
    const float* er = e + (r0 + i) * DIM;
    float a0 = b0, a1 = b1;
#pragma unroll
    for (int k = 0; k < DIM; ++k) {
      float ek = er[k];
      a0 = fmaf(ek, lw[d0 * DIM + k], a0);
      a1 = fmaf(ek, lw[d1 * DIM + k], a1);
    }
    s0 += tanhf(a0);
    s1 += tanhf(a1);
  }
  float* sps = acc + 6;
  atomicAdd(&sps[t * DIM + d0], s0 * (1.0f / N_NODES));
  atomicAdd(&sps[t * DIM + d1], s1 * (1.0f / N_NODES));
}

// ---------------- K5: sum of log(row/col exp-sums) per pair -----------------
__global__ __launch_bounds__(256) void k5_logred(const float* __restrict__ rc,
                                                 float* __restrict__ acc) {
  __shared__ float part[256];
  const int a = blockIdx.x;                 // 0..5
  const float* arr = rc + a * N_NODES;
  float s = 0.f;
  for (int i = threadIdx.x; i < N_NODES; i += 256) s += logf(arr[i]);
  part[threadIdx.x] = s;
  __syncthreads();
  for (int off = 128; off > 0; off >>= 1) {
    if (threadIdx.x < off) part[threadIdx.x] += part[threadIdx.x + off];
    __syncthreads();
  }
  if (threadIdx.x == 0) atomicAdd(&acc[3 + (a % 3)], part[0]);
}

// ---------------- K6: beta softmax, z_mc, loss scalar -----------------------
__global__ __launch_bounds__(256) void k6_final(
    const float* __restrict__ e0, const float* __restrict__ e1,
    const float* __restrict__ att_inter, const float* __restrict__ acc,
    float* __restrict__ out) {
  const float* sps = acc + 6;
  float s0 = 0.f, s1 = 0.f;
#pragma unroll
  for (int k = 0; k < DIM; ++k) {
    s0 += sps[k] * att_inter[k];
    s1 += sps[DIM + k] * att_inter[k];
  }
  float mm = fmaxf(s0, s1);
  float w0 = __expf(s0 - mm), w1 = __expf(s1 - mm);
  float beta0 = w0 / (w0 + w1), beta1 = 1.f - beta0;
  const int idx = blockIdx.x * 256 + threadIdx.x;
  out[idx] = beta0 * e0[idx] + beta1 * e1[idx];
  if (idx == 0) {
    // ntxent_p = (sum log rowsum + sum log colsum)/(2N) - dsum/N
    const float wts[3] = {1.f, 1.f, 0.5f};  // intra + ALPHA*inter
    float loss = 0.f;
#pragma unroll
    for (int pp = 0; pp < 3; ++pp)
      loss += wts[pp] * (acc[3 + pp] * (1.f / (2.f * N_NODES)) -
                         acc[pp] * (1.f / N_NODES));
    out[N_NODES * DIM] = loss;
  }
}

extern "C" void kernel_launch(void* const* d_in, const int* in_sizes, int n_in,
                              void* d_out, int out_size, void* d_ws,
                              size_t ws_size, hipStream_t stream) {
  const float* h0        = (const float*)d_in[0];
  const float* h1        = (const float*)d_in[1];
  const float* h2        = (const float*)d_in[2];
  const int*   nei0      = (const int*)d_in[3];
  const int*   nei1      = (const int*)d_in[4];
  const float* att0      = (const float*)d_in[5];
  const float* att1      = (const float*)d_in[6];
  const float* fc_w      = (const float*)d_in[7];
  const float* fc_b      = (const float*)d_in[8];
  const float* att_inter = (const float*)d_in[9];

  float* ws = (float*)d_ws;
  const int ND = N_NODES * DIM;                 // 524288
  float*    e0   = ws;                          // [ND]
  float*    e1   = ws + ND;                     // [ND]
  _Float16* a16  = (_Float16*)(ws + 2 * ND);    // [ND] halves
  _Float16* b16  = a16 + ND;
  _Float16* c16  = b16 + ND;
  float*    invn = ws + 2 * ND + (3 * ND) / 2;  // [3*8192]
  float*    rc   = invn + 3 * N_NODES;          // [6*8192] row/col exp-sums
  float*    acc  = rc + 6 * N_NODES;            // dsum[3], logacc[3], sps[128]

  // Zero accumulators (rc + acc) each call — graph-replay safe.
  hipMemsetAsync(rc, 0, (6 * N_NODES + 160) * sizeof(float), stream);

  k1_intra<<<dim3(N_NODES / 8, 2), 256, 0, stream>>>(h0, h1, h2, nei0, nei1,
                                                     att0, att1, e0, e1);
  k2_norm<<<dim3(N_NODES / 8, 3), 256, 0, stream>>>(h0, e0, e1, a16, b16, c16,
                                                    invn);
  k2b_diag<<<dim3(N_NODES / 8, 3), 256, 0, stream>>>(h0, e0, e1, invn, acc);
  k3_sim<<<dim3(N_NODES / 128, 8, 3), 256, 0, stream>>>(a16, b16, c16, rc);
  k4_fc<<<dim3(64, 2), 256, 0, stream>>>(e0, e1, fc_w, fc_b, acc);
  k5_logred<<<6, 256, 0, stream>>>(rc, acc);
  k6_final<<<ND / 256, 256, 0, stream>>>(e0, e1, att_inter, acc, (float*)d_out);
}
